// CausalAttention_12807592476992
// MI455X (gfx1250) — matrix operand order repaired and checked by reading, not compiled, not run on hardware
//
#include <hip/hip_runtime.h>
#include <hip/hip_bf16.h>

// ---------------------------------------------------------------------------
// CDNA5 (gfx1250) causal attention, bf16 WMMA, flash-style fused softmax.
// Wave32. WMMA shape 16x16x32 bf16 -> f32.
// ---------------------------------------------------------------------------

typedef __attribute__((ext_vector_type(16))) __bf16        v16bf;
typedef __attribute__((ext_vector_type(8)))  float         v8f;
typedef __attribute__((ext_vector_type(8)))  unsigned int  v8u;

#define BATCH 4
#define SEQ   2048
#define DMODEL 1024
#define MROWS (BATCH * SEQ)        // 8192 flattened rows

// fp32 -> bf16 round-to-nearest-even
static __device__ inline unsigned short f2bf(float f) {
  unsigned u = __builtin_bit_cast(unsigned, f);
  u += 0x7FFFu + ((u >> 16) & 1u);
  return (unsigned short)(u >> 16);
}
static __device__ inline unsigned pk2(unsigned short lo, unsigned short hi) {
  return (unsigned)lo | ((unsigned)hi << 16);
}
static __device__ inline v16bf make_frag(uint4 a, uint4 b) {
  v8u t;
  t[0] = a.x; t[1] = a.y; t[2] = a.z; t[3] = a.w;
  t[4] = b.x; t[5] = b.y; t[6] = b.z; t[7] = b.w;
  return __builtin_bit_cast(v16bf, t);
}
static __device__ inline v8f vzero8() {
  v8f z;
#pragma unroll
  for (int i = 0; i < 8; ++i) z[i] = 0.f;
  return z;
}

// ---------------------------------------------------------------------------
// Kernel 0: convert the three weight matrices fp32 -> bf16 ([3][1024][1024])
// ---------------------------------------------------------------------------
__global__ __launch_bounds__(256) void convw_kernel(
    const float* __restrict__ wq, const float* __restrict__ wk,
    const float* __restrict__ wv, unsigned short* __restrict__ Wb) {
  int idx = blockIdx.x * 256 + threadIdx.x;          // float4 index over 3M elems
  int z = idx >> 18;                                 // 262144 float4 per matrix
  int local = idx & 262143;
  const float* s = (z == 0) ? wq : (z == 1) ? wk : wv;
  float4 f = ((const float4*)s)[local];
  uint2 p;
  p.x = pk2(f2bf(f.x), f2bf(f.y));
  p.y = pk2(f2bf(f.z), f2bf(f.w));
  ((uint2*)Wb)[idx] = p;
}

// ---------------------------------------------------------------------------
// Kernel 1: QKV projection GEMM.  out = x @ W^T  (per z = q/k/v)
//   x fp32 staged & converted to bf16 in LDS (64 rows x 256-col chunks).
//   Wave tile 32(M) x 64(N) (2 A-frags share each B-frag -> half the W
//   traffic per WMMA); workgroup = 8 waves = 64(M) x 256(N).
//   z==0 -> Qb [8192][1024] bf16 row-major
//   z==1 -> Kb [8192][1024] bf16 row-major
//   z==2 -> Vt [1024][8192] bf16 (transposed: row = d, col = b*S+s)
// ---------------------------------------------------------------------------
#define XROWDW 132   // 128 data dwords (256 bf16) + 4 pad dwords per LDS row

__global__ __launch_bounds__(256) void qkv_kernel(
    const float* __restrict__ x, const unsigned short* __restrict__ Wb,
    unsigned short* __restrict__ Qb, unsigned short* __restrict__ Kb,
    unsigned short* __restrict__ Vt) {
  alignas(16) __shared__ unsigned int ldsX[64 * XROWDW];   // 64 rows x 256 bf16 chunk

  const int z     = blockIdx.z;
  const int mrow0 = blockIdx.y * 64;
  const int tid   = threadIdx.x;
  const int wid   = tid >> 5, lane = tid & 31;
  const int wm    = wid >> 2, wn   = wid & 3;
  const int hi    = lane >> 4, ln  = lane & 15;
  const int arow  = wm * 32 + ln;                       // LDS row (first M-tile)
  const int nb    = blockIdx.x * 256 + wn * 64;         // wave N base

  const unsigned short* W  = Wb + (size_t)z * DMODEL * DMODEL;
  const unsigned int*   Wd = (const unsigned int*)W;    // 512 dwords per W row

  v8f acc[2][4];
#pragma unroll
  for (int rt = 0; rt < 2; ++rt)
#pragma unroll
    for (int nt = 0; nt < 4; ++nt) acc[rt][nt] = vzero8();

  const float* xtile = x + (size_t)mrow0 * DMODEL;

  for (int ph = 0; ph < 4; ++ph) {                      // four 256-wide K chunks
    if (ph) __syncthreads();                            // previous chunk consumed
    // cooperative load 64x256 fp32 -> bf16 LDS
#pragma unroll
    for (int it = 0; it < 16; ++it) {
      int e = it * 1024 + tid * 4;                      // element in 64x256 chunk
      int r = e >> 8, c = e & 255;
      float4 f = *(const float4*)(xtile + (size_t)r * DMODEL + ph * 256 + c);
      unsigned* p = &ldsX[r * XROWDW + (c >> 1)];
      p[0] = pk2(f2bf(f.x), f2bf(f.y));
      p[1] = pk2(f2bf(f.z), f2bf(f.w));
    }
    __syncthreads();

    for (int ks = 0; ks < 8; ++ks) {                    // 8 K-steps of 32 per chunk
      // A fragments (16x32 bf16): low lanes dwords {0..3,8..11}, high {4..7,12..15}
      int adw = arow * XROWDW + ks * 16 + hi * 4;
      uint4 a0 = *(const uint4*)&ldsX[adw];
      uint4 a1 = *(const uint4*)&ldsX[adw + 8];
      v16bf af0 = make_frag(a0, a1);
      int adw2 = adw + 16 * XROWDW;                     // second M-tile (+16 rows)
      uint4 a2 = *(const uint4*)&ldsX[adw2];
      uint4 a3 = *(const uint4*)&ldsX[adw2 + 8];
      v16bf af1 = make_frag(a2, a3);
      int kglob = ph * 8 + ks;
#pragma unroll
      for (int nt = 0; nt < 4; ++nt) {
        // B fragment (32x16): lane = column n = W row; low lanes K-dwords {0..3,8..11}, high {4..7,12..15}
        int wrow = nb + nt * 16 + ln;
        size_t bdw = (size_t)wrow * 512 + kglob * 16 + hi * 4;
        uint4 b0 = *(const uint4*)(Wd + bdw);
        uint4 b1 = *(const uint4*)(Wd + bdw + 8);
        v16bf bf = make_frag(b0, b1);
        acc[0][nt] = __builtin_amdgcn_wmma_f32_16x16x32_bf16(
            false, af0, false, bf, (short)0, acc[0][nt], false, false);
        acc[1][nt] = __builtin_amdgcn_wmma_f32_16x16x32_bf16(
            false, af1, false, bf, (short)0, acc[1][nt], false, false);
      }
    }
  }

  // Epilogue: C layout -> lane holds N=ln, rows M = hi*8 + v
#pragma unroll
  for (int rt = 0; rt < 2; ++rt) {
    const int mbase = mrow0 + wm * 32 + rt * 16 + hi * 8;
    if (z < 2) {
      unsigned short* outp = (z == 0) ? Qb : Kb;
#pragma unroll
      for (int nt = 0; nt < 4; ++nt) {
        int n = nb + nt * 16 + ln;
#pragma unroll
        for (int v = 0; v < 8; ++v)
          outp[(size_t)(mbase + v) * DMODEL + n] = f2bf(acc[rt][nt][v]);
      }
    } else {
      // V transposed: Vt[n][m]; lane's 8 rows are contiguous -> one b128 store
#pragma unroll
      for (int nt = 0; nt < 4; ++nt) {
        int n = nb + nt * 16 + ln;
        uint4 pkv;
        pkv.x = pk2(f2bf(acc[rt][nt][0]), f2bf(acc[rt][nt][1]));
        pkv.y = pk2(f2bf(acc[rt][nt][2]), f2bf(acc[rt][nt][3]));
        pkv.z = pk2(f2bf(acc[rt][nt][4]), f2bf(acc[rt][nt][5]));
        pkv.w = pk2(f2bf(acc[rt][nt][6]), f2bf(acc[rt][nt][7]));
        *(uint4*)&Vt[(size_t)n * MROWS + mbase] = pkv;
      }
    }
  }
}

// ---------------------------------------------------------------------------
// Kernel 2: fused causal flash attention.
//   Workgroup: 16 query rows, 8 waves; wave w owns output d-chunk [w*128, w*128+128).
//   Per 32-key step: each wave -> 8 QK WMMAs (partial over its d-chunk),
//   LDS reduce across waves, online softmax (redundant in all waves),
//   wave0 publishes bf16 P tile, all waves -> 8 PV WMMAs against Vt.
//   Next step's K/V lines are prefetched (global_prefetch_b8) during softmax.
// ---------------------------------------------------------------------------
#define PROWDW 20    // P tile LDS row: 16 data dwords (32 bf16) + 4 pad

__global__ __launch_bounds__(256) void attn_kernel(
    const unsigned short* __restrict__ Qb, const unsigned short* __restrict__ Kb,
    const unsigned short* __restrict__ Vt, float* __restrict__ out) {
  alignas(32) __shared__ float        ldsS[8][2][32][8];       // 16 KB partial scores
  alignas(16) __shared__ unsigned int ldsP[16 * PROWDW];        // bf16 P tile 16x32

  const int tid = threadIdx.x, wid = tid >> 5, lane = tid & 31;
  const int hi = lane >> 4, ln = lane & 15;
  const int qbase = blockIdx.x * 16;
  const int b     = blockIdx.y;
  const int dbase = wid * 128;

  const unsigned int* Qd = (const unsigned int*)Qb;   // 512 dwords per row
  const unsigned int* Kd = (const unsigned int*)Kb;
  const unsigned int* Vd = (const unsigned int*)Vt;   // 4096 dwords per Vt row

  // Preload this wave's Q fragments (4 x 16x32 over its 128-wide d-chunk)
  const size_t qrow = (size_t)(b * SEQ + qbase + ln);
  v16bf qf[4];
#pragma unroll
  for (int kc = 0; kc < 4; ++kc) {
    size_t dw = qrow * 512 + (dbase >> 1) + kc * 16 + hi * 4;
    uint4 a0 = *(const uint4*)(Qd + dw);
    uint4 a1 = *(const uint4*)(Qd + dw + 8);
    qf[kc] = make_frag(a0, a1);
  }

  v8f acc[8];
#pragma unroll
  for (int nt = 0; nt < 8; ++nt) acc[nt] = vzero8();
  float mrow[8], lrow[8];
#pragma unroll
  for (int v = 0; v < 8; ++v) { mrow[v] = -3.0e38f; lrow[v] = 0.f; }

  const float scl = 0.03125f;                 // 1/sqrt(1024)
  const int nsteps = ((qbase + 15) >> 5) + 1; // causal bound, 32-key steps

  for (int kb = 0; kb < nsteps; ++kb) {
    const int kbase = kb * 32;

    // ---- partial scores over this wave's d-chunk: S = Q_chunk @ K_chunk^T
    v8f s0 = vzero8(), s1 = vzero8();
#pragma unroll
    for (int kc = 0; kc < 4; ++kc) {
      size_t k0 = (size_t)(b * SEQ + kbase + ln) * 512 + (dbase >> 1) + kc * 16 + hi * 4;
      size_t k1 = k0 + (size_t)16 * 512;      // second 16-key tile
      uint4 b0a = *(const uint4*)(Kd + k0);
      uint4 b0b = *(const uint4*)(Kd + k0 + 8);
      uint4 b1a = *(const uint4*)(Kd + k1);
      uint4 b1b = *(const uint4*)(Kd + k1 + 8);
      v16bf kf0 = make_frag(b0a, b0b);
      v16bf kf1 = make_frag(b1a, b1b);
      s0 = __builtin_amdgcn_wmma_f32_16x16x32_bf16(false, qf[kc], false, kf0,
                                                   (short)0, s0, false, false);
      s1 = __builtin_amdgcn_wmma_f32_16x16x32_bf16(false, qf[kc], false, kf1,
                                                   (short)0, s1, false, false);
    }
    *((v8f*)&ldsS[wid][0][lane][0]) = s0;
    *((v8f*)&ldsS[wid][1][lane][0]) = s1;

    // ---- prefetch next step's K and V lines (hides latency behind softmax)
    if (kb + 1 < nsteps) {
      size_t nk0 = (size_t)(b * SEQ + kbase + 32 + ln) * 512 + (dbase >> 1) + hi * 8;
      __builtin_prefetch(Kd + nk0, 0, 3);
      __builtin_prefetch(Kd + nk0 + (size_t)16 * 512, 0, 3);
#pragma unroll
      for (int nt = 0; nt < 8; ++nt) {
        int n = dbase + nt * 16 + ln;
        size_t nv = (size_t)n * (MROWS / 2) + ((b * SEQ + kbase + 32) >> 1) + hi * 8;
        __builtin_prefetch(Vd + nv, 0, 3);
      }
    }
    __syncthreads();

    // ---- reduce partials across the 8 waves (all waves, redundantly)
    v8f S0 = vzero8(), S1 = vzero8();
#pragma unroll
    for (int w = 0; w < 8; ++w) {
      S0 += *((const v8f*)&ldsS[w][0][lane][0]);
      S1 += *((const v8f*)&ldsS[w][1][lane][0]);
    }

    // ---- online softmax (row = hi*8+v across VGPR idx, cols across 16 lanes)
#pragma unroll
    for (int v = 0; v < 8; ++v) {
      int rg = qbase + hi * 8 + v;                       // global query row
      float a0 = S0[v] * scl, a1 = S1[v] * scl;
      if (kbase + ln > rg)      a0 = -3.0e38f;           // causal mask
      if (kbase + 16 + ln > rg) a1 = -3.0e38f;
      float mx = fmaxf(a0, a1);
#pragma unroll
      for (int off = 8; off > 0; off >>= 1)
        mx = fmaxf(mx, __shfl_xor(mx, off, 16));
      float mnew  = fmaxf(mrow[v], mx);
      float alpha = __expf(mrow[v] - mnew);
      float p0 = __expf(a0 - mnew);
      float p1 = __expf(a1 - mnew);
      float ps = p0 + p1;
#pragma unroll
      for (int off = 8; off > 0; off >>= 1)
        ps += __shfl_xor(ps, off, 16);
      lrow[v] = lrow[v] * alpha + ps;
      mrow[v] = mnew;
#pragma unroll
      for (int nt = 0; nt < 8; ++nt) acc[nt][v] *= alpha;  // rescale O
      if (wid == 0) {                                    // publish bf16 P tile
        unsigned short* pp = (unsigned short*)ldsP;
        int r = hi * 8 + v;
        pp[r * (2 * PROWDW) + ln]      = f2bf(p0);
        pp[r * (2 * PROWDW) + 16 + ln] = f2bf(p1);
      }
    }
    __syncthreads();

    // ---- PV: O_chunk += P(16x32) @ V(32 x 128)  via Vt (contiguous keys)
    int pdw = ln * PROWDW + hi * 4;
    uint4 p0 = *(const uint4*)&ldsP[pdw];
    uint4 p1 = *(const uint4*)&ldsP[pdw + 8];
    v16bf pf = make_frag(p0, p1);
#pragma unroll
    for (int nt = 0; nt < 8; ++nt) {
      int n = dbase + nt * 16 + ln;                      // Vt row = output dim
      size_t vdw = (size_t)n * (MROWS / 2) + ((b * SEQ + kbase) >> 1) + hi * 4;
      uint4 v0 = *(const uint4*)(Vd + vdw);
      uint4 v1 = *(const uint4*)(Vd + vdw + 8);
      v16bf vf = make_frag(v0, v1);
      acc[nt] = __builtin_amdgcn_wmma_f32_16x16x32_bf16(
          false, pf, false, vf, (short)0, acc[nt], false, false);
    }
    __syncthreads();                                     // protect ldsS/ldsP reuse
  }

  // ---- normalize and write fp32 output
#pragma unroll
  for (int nt = 0; nt < 8; ++nt) {
    int n = dbase + nt * 16 + ln;
#pragma unroll
    for (int v = 0; v < 8; ++v) {
      int rg = qbase + hi * 8 + v;
      out[(size_t)(b * SEQ + rg) * DMODEL + n] = acc[nt][v] / lrow[v];
    }
  }
}

// ---------------------------------------------------------------------------
// Launch: convw -> qkv (z = q/k/v) -> attn.   Workspace: 54 MB.
// ---------------------------------------------------------------------------
extern "C" void kernel_launch(void* const* d_in, const int* in_sizes, int n_in,
                              void* d_out, int out_size, void* d_ws, size_t ws_size,
                              hipStream_t stream) {
  const float* x  = (const float*)d_in[0];
  const float* wq = (const float*)d_in[1];
  const float* wk = (const float*)d_in[2];
  const float* wv = (const float*)d_in[3];

  unsigned short* Qb = (unsigned short*)d_ws;                  // [8192][1024] bf16
  unsigned short* Kb = Qb + (size_t)MROWS * DMODEL;            // [8192][1024] bf16
  unsigned short* Vt = Kb + (size_t)MROWS * DMODEL;            // [1024][8192] bf16
  unsigned short* Wb = Vt + (size_t)DMODEL * MROWS;            // [3][1024][1024] bf16

  convw_kernel<<<3 * DMODEL * DMODEL / 4 / 256, 256, 0, stream>>>(wq, wk, wv, Wb);

  dim3 g1(DMODEL / 256, MROWS / 64, 3);                        // (4, 128, 3)
  qkv_kernel<<<g1, 256, 0, stream>>>(x, Wb, Qb, Kb, Vt);

  dim3 g2(SEQ / 16, BATCH);                                    // (128, 4)
  attn_kernel<<<g2, 256, 0, stream>>>(Qb, Kb, Vt, (float*)d_out);
}
